// GNN_10797547782343
// MI455X (gfx1250) — compile-verified, matrix operand order checked by reading, not run on hardware
//
#include <hip/hip_runtime.h>

// ---------------------------------------------------------------------------
// Problem constants (match reference setup_inputs)
// ---------------------------------------------------------------------------
#define NN      100000
#define EE      400000
#define NEDGE   (EE + NN)      // edges + self loops
#define D_ATOM  98
#define D_BOND  13
#define EMB     128
#define HEADS   4
#define HE      (HEADS * EMB)  // 512
#define NLAYER  4

typedef __attribute__((ext_vector_type(16))) __bf16 v16bf;
typedef __attribute__((ext_vector_type(8)))  float  v8f;

union Frag { uint4 u[2]; v16bf v; };

__device__ __forceinline__ unsigned short f2bf(float f) {
    unsigned int u = __float_as_uint(f);
    unsigned int r = u + 0x7FFFu + ((u >> 16) & 1u);   // round-to-nearest-even
    return (unsigned short)(r >> 16);
}

__device__ __forceinline__ float preluf(float v, float a) {
    return v >= 0.0f ? v : a * v;
}

__device__ __forceinline__ void atomicMaxF(float* addr, float v) {
    // standard signed/unsigned bit-trick float max (works with -inf init)
    if (v >= 0.0f) atomicMax((int*)addr, __float_as_int(v));
    else           atomicMin((unsigned int*)addr, __float_as_uint(v));
}

// ---------------------------------------------------------------------------
// Utility fills / conversions / repacks
// ---------------------------------------------------------------------------
__global__ void fill_kernel(float* __restrict__ p, float v, long long n) {
    long long i = blockIdx.x * (long long)blockDim.x + threadIdx.x;
    if (i < n) p[i] = v;
}

// fp32 [M][Kin] -> bf16 [M][Kfull] zero-padded
__global__ void convert_pad_bf16(const float* __restrict__ X,
                                 unsigned short* __restrict__ Xb,
                                 int M, int Kin, int Kfull) {
    long long i = blockIdx.x * (long long)blockDim.x + threadIdx.x;
    if (i >= (long long)M * Kfull) return;
    int c = (int)(i % Kfull);
    long long r = i / Kfull;
    Xb[i] = f2bf(c < Kin ? X[r * Kin + c] : 0.0f);
}

// Repack row-major fp32 weight [Krows][Nout] into WMMA-B fragment layout:
// tile t = nt*KT + kt ; per lane 16 bf16 halves matching 16-bit B frag layout.
__global__ void repack_w_kernel(const float* __restrict__ W,
                                unsigned short* __restrict__ Wp,
                                int Krows, int Kfull, int Nout) {
    int KT = Kfull >> 5, NT = Nout >> 4;
    int idx = blockIdx.x * blockDim.x + threadIdx.x;   // tile*32 + lane
    int total = KT * NT * 32;
    if (idx >= total) return;
    int lane = idx & 31;
    int t = idx >> 5;
    int kt = t % KT, nt = t / KT;
    int n = nt * 16 + (lane & 15);
    unsigned short* dst = Wp + (size_t)idx * 16;
    #pragma unroll
    for (int h = 0; h < 16; ++h) {
        int k = kt * 32 + ((lane >> 4) * 8) + ((h >> 3) * 16) + (h & 7);
        float v = (k < Krows) ? W[(size_t)k * Nout + n] : 0.0f;
        dst[h] = f2bf(v);
    }
}

// ee_W [L][13][512] -> [L][512][13] (contiguous per output column)
__global__ void transpose_eew(const float* __restrict__ W,
                              float* __restrict__ Wt) {
    int idx = blockIdx.x * blockDim.x + threadIdx.x;
    if (idx >= NLAYER * HE * D_BOND) return;
    int k = idx % D_BOND;
    int c = (idx / D_BOND) % HE;
    int l = idx / (D_BOND * HE);
    Wt[idx] = W[(size_t)l * D_BOND * HE + (size_t)k * HE + c];
}

// ---------------------------------------------------------------------------
// WMMA bf16 GEMM:  out = prelu(A @ B + bias)
//   A : [M][Kfull] bf16 row-major, Bp : fragment-packed, one 16x16 tile/wave.
//   Writes fp32 (outF) and/or bf16 (outBF).
// ---------------------------------------------------------------------------
__global__ void wmma_gemm_kernel(const unsigned short* __restrict__ A,
                                 const unsigned short* __restrict__ Bp,
                                 const float* __restrict__ bias,   // may be null
                                 const float* __restrict__ prelu_a,
                                 float* __restrict__ outF,          // may be null
                                 unsigned short* __restrict__ outBF,// may be null
                                 int M, int Kfull, int Nout) {
    const int KT = Kfull >> 5;
    const int NT = Nout >> 4;
    const int Mtiles = M >> 4;
    int wid = blockIdx.x * (blockDim.x >> 5) + (threadIdx.x >> 5);
    if (wid >= Mtiles * NT) return;
    int nt = wid % NT;
    int mt = wid / NT;
    int lane = threadIdx.x & 31;
    int row = mt * 16 + (lane & 15);
    int khalf = (lane >> 4) * 8;

    const unsigned short* arow = A + (size_t)row * Kfull + khalf;
    const unsigned short* bptr = Bp + ((size_t)(nt * KT) * 32 + lane) * 16;

    v8f c = {};
    for (int kt = 0; kt < KT; ++kt) {
        Frag a, b;
        a.u[0] = *(const uint4*)(arow);
        a.u[1] = *(const uint4*)(arow + 16);   // K + 16
        b.u[0] = *(const uint4*)(bptr);
        b.u[1] = *(const uint4*)(bptr + 8);
        __builtin_prefetch(arow + 32, 0, 1);   // global_prefetch_b8 of next k-slab
        c = __builtin_amdgcn_wmma_f32_16x16x32_bf16(
                false, a.v, false, b.v, (short)0, c, false, false);
        arow += 32;
        bptr += 32 * 16;
    }

    float a_slope = *prelu_a;
    int ncol = nt * 16 + (lane & 15);
    float bv = bias ? bias[ncol] : 0.0f;
    int mbase = mt * 16 + (lane >> 4) * 8;
    #pragma unroll
    for (int r = 0; r < 8; ++r) {
        float v = preluf(c[r] + bv, a_slope);
        size_t o = (size_t)(mbase + r) * Nout + ncol;
        if (outF)  outF[o]  = v;
        if (outBF) outBF[o] = f2bf(v);
    }
}

// ---------------------------------------------------------------------------
// Per-node attention reductions: ai[n,h] = xw[n,h*128:] . att1, aj with att2
// ---------------------------------------------------------------------------
__global__ void ai_aj_kernel(const float* __restrict__ xw,
                             const float* __restrict__ att,  // + l*1024
                             float* __restrict__ ai, float* __restrict__ aj) {
    int idx = blockIdx.x * blockDim.x + threadIdx.x;  // n*4 + h
    if (idx >= NN * HEADS) return;
    int h = idx & 3;
    int n = idx >> 2;
    const float* xr = xw + (size_t)n * HE + h * EMB;
    const float* a1 = att + h * 2 * EMB;
    const float* a2 = a1 + EMB;
    float s1 = 0.0f, s2 = 0.0f;
    #pragma unroll 4
    for (int c = 0; c < EMB; c += 4) {
        float4 x  = *(const float4*)(xr + c);
        float4 t1 = *(const float4*)(a1 + c);
        float4 t2 = *(const float4*)(a2 + c);
        s1 += x.x * t1.x + x.y * t1.y + x.z * t1.z + x.w * t1.w;
        s2 += x.x * t2.x + x.y * t2.y + x.z * t2.z + x.w * t2.w;
    }
    ai[idx] = s1;
    aj[idx] = s2;
}

// ---------------------------------------------------------------------------
// Edge pass 1: alpha logits (recomputes ee on the fly), fused segment-max(src)
// One wave per edge; lane covers 16 of the 512 columns.
// ---------------------------------------------------------------------------
__global__ void edge_alpha_kernel(const long long* __restrict__ eidx,
                                  const float* __restrict__ eattr,
                                  const float* __restrict__ eeWt,  // [512][13]
                                  const float* __restrict__ eeb,   // [512]
                                  const float* __restrict__ att,   // + l*1024
                                  const float* __restrict__ ai,
                                  const float* __restrict__ aj,
                                  const float* __restrict__ prelu_a,
                                  float* __restrict__ alpha,
                                  float* __restrict__ segmax) {
    int wid = blockIdx.x * (blockDim.x >> 5) + (threadIdx.x >> 5);
    if (wid >= NEDGE) return;
    int lane = threadIdx.x & 31;

    int s, d;
    float eav[D_BOND];
    if (wid < EE) {
        s = (int)eidx[wid];
        d = (int)eidx[EE + wid];
        const float* er = eattr + (size_t)wid * D_BOND;
        #pragma unroll
        for (int k = 0; k < D_BOND; ++k) eav[k] = er[k];
    } else {
        s = d = wid - EE;
        #pragma unroll
        for (int k = 0; k < D_BOND; ++k) eav[k] = 0.0f;
    }

    float a_slope = *prelu_a;
    float acc[HEADS] = {0.f, 0.f, 0.f, 0.f};
    #pragma unroll 4
    for (int j = 0; j < 16; ++j) {
        int c = j * 32 + lane;
        const float* wr = eeWt + c * D_BOND;
        float v = eeb[c];
        #pragma unroll
        for (int k = 0; k < D_BOND; ++k) v += eav[k] * wr[k];
        v = preluf(v, a_slope);
        int h = c >> 7;
        acc[h] += v * att[h * 2 * EMB + EMB + (c & 127)];
    }
    #pragma unroll
    for (int off = 16; off > 0; off >>= 1) {
        #pragma unroll
        for (int h = 0; h < HEADS; ++h)
            acc[h] += __shfl_down(acc[h], off, 32);
    }
    if (lane == 0) {
        #pragma unroll
        for (int h = 0; h < HEADS; ++h) {
            float lg = acc[h] + ai[d * HEADS + h] + aj[s * HEADS + h];
            lg = preluf(lg, a_slope);
            alpha[(size_t)wid * HEADS + h] = lg;
            atomicMaxF(&segmax[s * HEADS + h], lg);
        }
    }
}

// ---------------------------------------------------------------------------
// Edge pass 2: alpha = exp(alpha - max[src]); segment-sum(src)
// ---------------------------------------------------------------------------
__global__ void edge_exp_kernel(const long long* __restrict__ eidx,
                                const float* __restrict__ segmax,
                                float* __restrict__ alpha,
                                float* __restrict__ ssum) {
    int e = blockIdx.x * blockDim.x + threadIdx.x;
    if (e >= NEDGE) return;
    int s = (e < EE) ? (int)eidx[e] : (e - EE);
    #pragma unroll
    for (int h = 0; h < HEADS; ++h) {
        float v = __expf(alpha[(size_t)e * HEADS + h] - segmax[s * HEADS + h]);
        alpha[(size_t)e * HEADS + h] = v;
        atomicAdd(&ssum[s * HEADS + h], v);
    }
}

// ---------------------------------------------------------------------------
// Edge pass 3: msg = (xw[src]+ee) * alpha/(sum+eps), scatter-add to dst
// ---------------------------------------------------------------------------
__global__ void edge_msg_kernel(const long long* __restrict__ eidx,
                                const float* __restrict__ eattr,
                                const float* __restrict__ eeWt,
                                const float* __restrict__ eeb,
                                const float* __restrict__ prelu_a,
                                const float* __restrict__ alpha,
                                const float* __restrict__ ssum,
                                const float* __restrict__ xw,
                                float* __restrict__ aggr) {
    int wid = blockIdx.x * (blockDim.x >> 5) + (threadIdx.x >> 5);
    if (wid >= NEDGE) return;
    int lane = threadIdx.x & 31;

    int s, d;
    float eav[D_BOND];
    if (wid < EE) {
        s = (int)eidx[wid];
        d = (int)eidx[EE + wid];
        const float* er = eattr + (size_t)wid * D_BOND;
        #pragma unroll
        for (int k = 0; k < D_BOND; ++k) eav[k] = er[k];
    } else {
        s = d = wid - EE;
        #pragma unroll
        for (int k = 0; k < D_BOND; ++k) eav[k] = 0.0f;
    }

    float w[HEADS];
    #pragma unroll
    for (int h = 0; h < HEADS; ++h)
        w[h] = alpha[(size_t)wid * HEADS + h] /
               (ssum[s * HEADS + h] + 1e-16f);

    float a_slope = *prelu_a;
    #pragma unroll 4
    for (int j = 0; j < 16; ++j) {
        int c = j * 32 + lane;
        const float* wr = eeWt + c * D_BOND;
        float v = eeb[c];
        #pragma unroll
        for (int k = 0; k < D_BOND; ++k) v += eav[k] * wr[k];
        v = preluf(v, a_slope);
        float xj = xw[(size_t)s * HE + c] + v;
        atomicAdd(&aggr[(size_t)d * HE + c], xj * w[c >> 7]);
    }
}

// ---------------------------------------------------------------------------
// Head-mean + bias + LayerNorm + (PReLU | output). Block = 128 thr per node.
// ---------------------------------------------------------------------------
__global__ void final_ln_kernel(const float* __restrict__ aggr,
                                const float* __restrict__ gbias,
                                const float* __restrict__ lng,
                                const float* __restrict__ lnb,
                                const float* __restrict__ prelu_gnn,
                                float* __restrict__ outF,
                                unsigned short* __restrict__ hbf,
                                int is_last) {
    __shared__ float red[EMB];
    int n = blockIdx.x;
    int c = threadIdx.x;
    const float* ar = aggr + (size_t)n * HE;
    float v = 0.25f * (ar[c] + ar[EMB + c] + ar[2 * EMB + c] + ar[3 * EMB + c])
              + gbias[c];
    red[c] = v;
    __syncthreads();
    for (int off = 64; off > 0; off >>= 1) {
        if (c < off) red[c] += red[c + off];
        __syncthreads();
    }
    float mu = red[0] * (1.0f / EMB);
    __syncthreads();
    float dv = v - mu;
    red[c] = dv * dv;
    __syncthreads();
    for (int off = 64; off > 0; off >>= 1) {
        if (c < off) red[c] += red[c + off];
        __syncthreads();
    }
    float var = red[0] * (1.0f / EMB);
    float hn = dv * rsqrtf(var + 1e-5f) * lng[c] + lnb[c];
    if (is_last) {
        outF[(size_t)n * EMB + c] = hn;
    } else {
        float a = *prelu_gnn;
        hbf[(size_t)n * EMB + c] = f2bf(preluf(hn, a));
    }
}

// ---------------------------------------------------------------------------
// Host side
// ---------------------------------------------------------------------------
extern "C" void kernel_launch(void* const* d_in, const int* in_sizes, int n_in,
                              void* d_out, int out_size, void* d_ws, size_t ws_size,
                              hipStream_t stream) {
    (void)in_sizes; (void)n_in; (void)out_size; (void)ws_size;

    const float*     x        = (const float*)d_in[0];
    const long long* eidx     = (const long long*)d_in[1];
    const float*     eattr    = (const float*)d_in[2];
    const float*     xembW    = (const float*)d_in[3];
    const float*     pr_gnn   = (const float*)d_in[4];
    const float*     wlW      = (const float*)d_in[5];
    const float*     wlb      = (const float*)d_in[6];
    const float*     att      = (const float*)d_in[7];
    const float*     gbias    = (const float*)d_in[8];
    const float*     eeW      = (const float*)d_in[9];
    const float*     eeb      = (const float*)d_in[10];
    const float*     pr_gat   = (const float*)d_in[11];
    const float*     lng      = (const float*)d_in[12];
    const float*     lnb      = (const float*)d_in[13];
    float*           out      = (float*)d_out;

    // --- workspace carve-up (aligned to 256B) ---
    char* base = (char*)d_ws;
    size_t off = 0;
    auto alloc = [&](size_t nbytes) -> void* {
        void* p = base + off;
        off = (off + nbytes + 255) & ~(size_t)255;
        return p;
    };
    unsigned short* xbf  = (unsigned short*)alloc((size_t)NN * EMB * 2);
    unsigned short* hbf  = (unsigned short*)alloc((size_t)NN * EMB * 2);
    unsigned short* wpE  = (unsigned short*)alloc((size_t)EMB * EMB * 2);
    unsigned short* wpL  = (unsigned short*)alloc((size_t)NLAYER * EMB * HE * 2);
    float* eeWt  = (float*)alloc((size_t)NLAYER * HE * D_BOND * 4);
    float* xw    = (float*)alloc((size_t)NN * HE * 4);
    float* aggr  = (float*)alloc((size_t)NN * HE * 4);
    float* ai    = (float*)alloc((size_t)NN * HEADS * 4);
    float* aj    = (float*)alloc((size_t)NN * HEADS * 4);
    float* alpha = (float*)alloc((size_t)NEDGE * HEADS * 4);
    float* segm  = (float*)alloc((size_t)NN * HEADS * 4);
    float* ssum  = (float*)alloc((size_t)NN * HEADS * 4);

    const int T = 256;
    auto cdiv = [](long long a, long long b) -> int { return (int)((a + b - 1) / b); };

    // --- one-time preprocessing ---
    convert_pad_bf16<<<cdiv((long long)NN * EMB, T), T, 0, stream>>>(
        x, xbf, NN, D_ATOM, EMB);
    repack_w_kernel<<<cdiv((EMB / 32) * (EMB / 16) * 32, T), T, 0, stream>>>(
        xembW, wpE, D_ATOM, EMB, EMB);
    for (int l = 0; l < NLAYER; ++l)
        repack_w_kernel<<<cdiv((EMB / 32) * (HE / 16) * 32, T), T, 0, stream>>>(
            wlW + (size_t)l * EMB * HE, wpL + (size_t)l * EMB * HE, EMB, EMB, HE);
    transpose_eew<<<cdiv(NLAYER * HE * D_BOND, T), T, 0, stream>>>(eeW, eeWt);

    // --- input embedding: h = prelu(x @ x_emb_W) -> bf16 ---
    {
        int waves = (NN / 16) * (EMB / 16);
        wmma_gemm_kernel<<<cdiv(waves, T / 32), T, 0, stream>>>(
            xbf, wpE, nullptr, pr_gnn, nullptr, hbf, NN, EMB, EMB);
    }

    // --- GAT layers ---
    for (int l = 0; l < NLAYER; ++l) {
        fill_kernel<<<cdiv((long long)NN * HE, T), T, 0, stream>>>(
            aggr, 0.0f, (long long)NN * HE);
        fill_kernel<<<cdiv((long long)NN * HEADS, T), T, 0, stream>>>(
            segm, -INFINITY, (long long)NN * HEADS);
        fill_kernel<<<cdiv((long long)NN * HEADS, T), T, 0, stream>>>(
            ssum, 0.0f, (long long)NN * HEADS);

        // xw = prelu(h @ W + b)
        int waves = (NN / 16) * (HE / 16);
        wmma_gemm_kernel<<<cdiv(waves, T / 32), T, 0, stream>>>(
            hbf, wpL + (size_t)l * EMB * HE, wlb + (size_t)l * HE,
            pr_gat + l, xw, nullptr, NN, EMB, HE);

        ai_aj_kernel<<<cdiv((long long)NN * HEADS, T), T, 0, stream>>>(
            xw, att + (size_t)l * HEADS * 2 * EMB, ai, aj);

        int eblocks = cdiv((long long)NEDGE * 32, T);  // wave per edge
        edge_alpha_kernel<<<eblocks, T, 0, stream>>>(
            eidx, eattr, eeWt + (size_t)l * HE * D_BOND, eeb + (size_t)l * HE,
            att + (size_t)l * HEADS * 2 * EMB, ai, aj, pr_gat + l, alpha, segm);

        edge_exp_kernel<<<cdiv(NEDGE, T), T, 0, stream>>>(
            eidx, segm, alpha, ssum);

        edge_msg_kernel<<<eblocks, T, 0, stream>>>(
            eidx, eattr, eeWt + (size_t)l * HE * D_BOND, eeb + (size_t)l * HE,
            pr_gat + l, alpha, ssum, xw, aggr);

        final_ln_kernel<<<NN, EMB, 0, stream>>>(
            aggr, gbias + (size_t)l * EMB, lng + (size_t)l * EMB,
            lnb + (size_t)l * EMB, pr_gnn, out, hbf, (l == NLAYER - 1) ? 1 : 0);
    }
}
